// EncoderSupernodes_50130858278963
// MI455X (gfx1250) — matrix operand order repaired and checked by reading, not compiled, not run on hardware
//
#include <hip/hip_runtime.h>
#include <hip/hip_bf16.h>
#include <math.h>

typedef _Float16 h16;
typedef __attribute__((ext_vector_type(16))) _Float16 v16h;
typedef __attribute__((ext_vector_type(8)))  _Float16 v8h;
typedef __attribute__((ext_vector_type(8)))  float    v8f;

#define NB    4
#define NPTS  32768
#define NS    2048
#define KNBR  32
#define GNN   256
#define ENC   384
#define NHEAD 6
#define HD    64
#define DEPTH 4
#define TOK   (NB*NS)          // 8192
#define MLPH  (4*ENC)          // 1536

// ---------------------------------------------------------------- helpers
__device__ __forceinline__ v16h cat8(v8h lo, v8h hi) {
    return __builtin_shufflevector(lo, hi, 0,1,2,3,4,5,6,7,8,9,10,11,12,13,14,15);
}
// A fragment (16x32 f16, M x K) from row-major f16, row stride lda (halves).
// ISA layout: lane L holds row L%16; K(e) = (e%8) + 16*(e/8) + 8*(L/16)
__device__ __forceinline__ v16h ldA(const h16* A, int lda, int row, int k0, int lane) {
    int g = lane >> 4;
    const h16* p = A + (size_t)row * lda + k0 + 8 * g;
    v8h lo = *(const v8h*)p;
    v8h hi = *(const v8h*)(p + 16);
    return cat8(lo, hi);
}
// B fragment (32x16, K x N) read from BT stored N x K row-major (contiguous K).
// lane L holds column L%16; b[e] = B[k0 + (e%8) + 16*(e/8) + 8*(L/16)][col]
__device__ __forceinline__ v16h ldBT(const h16* BT, int ldb, int col, int k0, int lane) {
    int g = lane >> 4;
    const h16* p = BT + (size_t)col * ldb + k0 + 8 * g;
    v8h lo = *(const v8h*)p;
    v8h hi = *(const v8h*)(p + 16);
    return cat8(lo, hi);
}
__device__ __forceinline__ float gelu_f(float x) {
    float u = 0.7978845608028654f * (x + 0.044715f * x * x * x);
    float t = 1.f - 2.f / (__expf(2.f * u) + 1.f);   // tanh(u)
    return 0.5f * x * (1.f + t);
}
__device__ __forceinline__ float red32_sum(float v) {
    v += __shfl_xor(v, 16, 32); v += __shfl_xor(v, 8, 32);
    v += __shfl_xor(v, 4, 32);  v += __shfl_xor(v, 2, 32);
    v += __shfl_xor(v, 1, 32);  return v;
}
__device__ __forceinline__ float red16_max(float v) {
    v = fmaxf(v, __shfl_xor(v, 8, 32)); v = fmaxf(v, __shfl_xor(v, 4, 32));
    v = fmaxf(v, __shfl_xor(v, 2, 32)); v = fmaxf(v, __shfl_xor(v, 1, 32));
    return v;
}
__device__ __forceinline__ float red16_sum(float v) {
    v += __shfl_xor(v, 8, 32); v += __shfl_xor(v, 4, 32);
    v += __shfl_xor(v, 2, 32); v += __shfl_xor(v, 1, 32);
    return v;
}

// ---------------------------------------------------------------- weight prep
// WT[n*K + k] = (h16) W[k*N + n]
__global__ void wprep_kernel(const float* __restrict__ W, h16* __restrict__ WT,
                             int Kd, int N) {
    int idx = blockIdx.x * 256 + threadIdx.x;
    if (idx < Kd * N) {
        int k = idx / N, n = idx % N;
        WT[(size_t)n * Kd + k] = (h16)W[idx];
    }
}

// ---------------------------------------------------------------- KNN (one wave / supernode)
#define KNN_WAVES 4
__global__ void __launch_bounds__(128) knn_kernel(const float* __restrict__ points,
                                                  const int* __restrict__ snidx,
                                                  float* __restrict__ nbr_xyz) {
    __shared__ float cd[KNN_WAVES][1024];
    __shared__ int   ci[KNN_WAVES][1024];
    int wave = threadIdx.x >> 5, lane = threadIdx.x & 31;
    int sid = blockIdx.x * KNN_WAVES + wave;         // 0 .. 8191
    int b = sid / NS, s = sid % NS;
    const float* pb = points + (size_t)b * NPTS * 3;
    int sn = snidx[b * NS + s];
    float sx = pb[sn * 3 + 0], sy = pb[sn * 3 + 1], sz = pb[sn * 3 + 2];

    float bd[KNBR]; int bi[KNBR];
#pragma unroll
    for (int j = 0; j < KNBR; ++j) { bd[j] = 3.4e38f; bi[j] = -1; }
    float cmax = 3.4e38f;
    for (int i = 0; i < NPTS / 32; ++i) {
        int n = lane + 32 * i;
        float dx = pb[n * 3 + 0] - sx, dy = pb[n * 3 + 1] - sy, dz = pb[n * 3 + 2] - sz;
        float d = dx * dx + dy * dy + dz * dz;
        if (d < cmax) {
            bool done = false;
#pragma unroll
            for (int j = 0; j < KNBR; ++j) {
                bool rep = (!done) && (bd[j] == cmax);
                if (rep) { bd[j] = d; bi[j] = n; done = true; }
            }
            cmax = -1.f;
#pragma unroll
            for (int j = 0; j < KNBR; ++j) cmax = fmaxf(cmax, bd[j]);
        }
    }
#pragma unroll
    for (int j = 0; j < KNBR; ++j) { cd[wave][lane * KNBR + j] = bd[j]; ci[wave][lane * KNBR + j] = bi[j]; }
    __syncthreads();
    int rank[KNBR];
#pragma unroll
    for (int j = 0; j < KNBR; ++j) rank[j] = 0;
    for (int m = 0; m < 1024; ++m) {            // LDS broadcast read
        float dm = cd[wave][m]; int im = ci[wave][m];
#pragma unroll
        for (int j = 0; j < KNBR; ++j)
            rank[j] += ((dm < bd[j]) || (dm == bd[j] && im < bi[j])) ? 1 : 0;
    }
    float* outp = nbr_xyz + (size_t)sid * KNBR * 3;
#pragma unroll
    for (int j = 0; j < KNBR; ++j) {
        if (rank[j] < KNBR) {
            int n = bi[j];
            outp[rank[j] * 3 + 0] = pb[n * 3 + 0];
            outp[rank[j] * 3 + 1] = pb[n * 3 + 1];
            outp[rank[j] * 3 + 2] = pb[n * 3 + 2];
        }
    }
}

// ---------------------------------------------------------------- fused GNN stage (one block / supernode)
__global__ void __launch_bounds__(128) gnn_kernel(const float* __restrict__ nbr_xyz,
                                                  const float* __restrict__ pool_w,
                                                  const float* __restrict__ pool_b,
                                                  const h16* __restrict__ w1T, const float* __restrict__ b1,
                                                  const h16* __restrict__ w2T, const float* __restrict__ b2,
                                                  h16* __restrict__ pooled16) {
    __shared__ alignas(16) h16 x0[KNBR][GNN];   // 16 KB
    __shared__ alignas(16) h16 h1[KNBR][GNN];   // 16 KB
    __shared__ float psum[GNN];                 // 1 KB
    int sid = blockIdx.x;
    int tid = threadIdx.x, wave = tid >> 5, lane = tid & 31;
    const float* nb = nbr_xyz + (size_t)sid * KNBR * 3;
    const float EMB_NEG = -0.22464244810917523f;   // -ln(10000)/41

    for (int i = tid; i < GNN; i += 128) psum[i] = 0.f;
    for (int idx = tid; idx < KNBR * GNN; idx += 128) {
        int n = idx >> 8, c = idx & 255;
        float px = nb[n * 3 + 0], py = nb[n * 3 + 1], pz = nb[n * 3 + 2];
        float v = pool_b[c] + px * pool_w[0 * GNN + c] + py * pool_w[1 * GNN + c] + pz * pool_w[2 * GNN + c];
        if (c < 252) {
            int d = c / 84, r = c % 84;
            int j = (r < 42) ? r : r - 42;
            float pd = (d == 0) ? px : ((d == 1) ? py : pz);
            float ang = pd * __expf(EMB_NEG * (float)j);
            v += (r < 42) ? __sinf(ang) : __cosf(ang);
        }
        x0[n][c] = (h16)v;
    }
    __syncthreads();
    // GEMM1: (32x256) @ (256x256) + b1, gelu -> h1
    for (int i = 0; i < 8; ++i) {
        int tile = i * 4 + wave; int mt = tile >> 4, nt = tile & 15;
        v8f acc = {};
#pragma unroll
        for (int k0 = 0; k0 < GNN; k0 += 32) {
            v16h a = ldA(&x0[0][0], GNN, mt * 16 + (lane & 15), k0, lane);
            v16h bf = ldBT(w1T, GNN, nt * 16 + (lane & 15), k0, lane);
            acc = __builtin_amdgcn_wmma_f32_16x16x32_f16(false, a, false, bf, (short)0, acc, false, false);
        }
        int col = nt * 16 + (lane & 15);
        int rb = mt * 16 + 8 * (lane >> 4);
        float bias = b1[col];
#pragma unroll
        for (int v = 0; v < 8; ++v) h1[rb + v][col] = (h16)gelu_f(acc[v] + bias);
    }
    __syncthreads();
    // GEMM2: (32x256) @ (256x256); column sums via LDS atomics (mean pool)
    for (int i = 0; i < 8; ++i) {
        int tile = i * 4 + wave; int mt = tile >> 4, nt = tile & 15;
        v8f acc = {};
#pragma unroll
        for (int k0 = 0; k0 < GNN; k0 += 32) {
            v16h a = ldA(&h1[0][0], GNN, mt * 16 + (lane & 15), k0, lane);
            v16h bf = ldBT(w2T, GNN, nt * 16 + (lane & 15), k0, lane);
            acc = __builtin_amdgcn_wmma_f32_16x16x32_f16(false, a, false, bf, (short)0, acc, false, false);
        }
        int col = nt * 16 + (lane & 15);
        float part = 0.f;
#pragma unroll
        for (int v = 0; v < 8; ++v) part += acc[v];
        atomicAdd(&psum[col], part);
    }
    __syncthreads();
    for (int c = tid; c < GNN; c += 128)
        pooled16[(size_t)sid * GNN + c] = (h16)(psum[c] * (1.f / KNBR) + b2[c]);
}

// ---------------------------------------------------------------- generic WMMA GEMM
// out = act( A(f16,MxK) @ BT(f16,NxK)^T + bias ) [+ resid]; per-wave 16x64 strip
__global__ void __launch_bounds__(256) gemm_kernel(const h16* __restrict__ A,
                                                   const h16* __restrict__ BT,
                                                   const float* __restrict__ bias,
                                                   float* __restrict__ out32,
                                                   h16* __restrict__ out16,
                                                   const float* __restrict__ resid,
                                                   int M, int N, int Kd, int act) {
    int wave = threadIdx.x >> 5, lane = threadIdx.x & 31;
    int m0 = blockIdx.y * 128 + wave * 16;
    int n0 = blockIdx.x * 64;
    int cl = lane & 15, g = lane >> 4;
    const h16* arow = A + (size_t)(m0 + cl) * Kd;
    v8f acc[4] = {{}, {}, {}, {}};
    for (int k0 = 0; k0 < Kd; k0 += 32) {
        const h16* p = arow + k0 + 8 * g;
        v16h a = cat8(*(const v8h*)p, *(const v8h*)(p + 16));
#pragma unroll
        for (int t = 0; t < 4; ++t) {
            v16h bf = ldBT(BT, Kd, n0 + t * 16 + cl, k0, lane);
            acc[t] = __builtin_amdgcn_wmma_f32_16x16x32_f16(false, a, false, bf, (short)0, acc[t], false, false);
        }
    }
#pragma unroll
    for (int t = 0; t < 4; ++t) {
        int col = n0 + t * 16 + cl;
        float bb = bias ? bias[col] : 0.f;
#pragma unroll
        for (int v = 0; v < 8; ++v) {
            int r = m0 + v + 8 * g;
            float val = acc[t][v] + bb;
            if (act) val = gelu_f(val);
            size_t o = (size_t)r * N + col;
            if (resid) val += resid[o];
            if (out32) out32[o] = val;
            if (out16) out16[o] = (h16)val;
        }
    }
}

// ---------------------------------------------------------------- LayerNorm (one wave / row)
__global__ void __launch_bounds__(256) ln_kernel(const float* __restrict__ x,
                                                 const float* __restrict__ sc,
                                                 const float* __restrict__ bt,
                                                 h16* __restrict__ out) {
    int wave = threadIdx.x >> 5, lane = threadIdx.x & 31;
    int row = blockIdx.x * 8 + wave;
    const float* xr = x + (size_t)row * ENC;
    float v[12]; float sum = 0.f;
#pragma unroll
    for (int i = 0; i < 12; ++i) { v[i] = xr[lane + 32 * i]; sum += v[i]; }
    sum = red32_sum(sum);
    float mu = sum * (1.f / ENC);
    float vs = 0.f;
#pragma unroll
    for (int i = 0; i < 12; ++i) { float d = v[i] - mu; vs += d * d; }
    vs = red32_sum(vs);
    float inv = rsqrtf(vs * (1.f / ENC) + 1e-6f);
#pragma unroll
    for (int i = 0; i < 12; ++i) {
        int c = lane + 32 * i;
        out[(size_t)row * ENC + c] = (h16)((v[i] - mu) * inv * sc[c] + bt[c]);
    }
}

// ---------------------------------------------------------------- V transpose: (tok, h, d) -> (b, h, d, t)
__global__ void vtr_kernel(const h16* __restrict__ vsrc, h16* __restrict__ vt) {
    int idx = blockIdx.x * 256 + threadIdx.x;
    if (idx >= TOK * ENC) return;
    int t = idx / ENC, c = idx % ENC;
    int b = t / NS, tt = t % NS;
    int h = c / HD, d = c % HD;
    vt[(((size_t)b * NHEAD + h) * HD + d) * NS + tt] = vsrc[idx];
}

// ---------------------------------------------------------------- flash attention (one wave / 16-query tile)
__global__ void __launch_bounds__(128) attn_kernel(const h16* __restrict__ q16,
                                                   const h16* __restrict__ k16,
                                                   const h16* __restrict__ vt16,
                                                   h16* __restrict__ o16) {
    __shared__ alignas(16) h16 ptile[4][16][32];   // 4 KB, per-wave slice
    int wave = threadIdx.x >> 5, lane = threadIdx.x & 31;
    int g = lane >> 4, cl = lane & 15;
    int h = blockIdx.y, b = blockIdx.z;
    int qt0 = (blockIdx.x * 4 + wave) * 16;
    const h16* qb = q16 + ((size_t)(b * NS + qt0)) * ENC + h * HD;
    const h16* kb = k16 + ((size_t)(b * NS)) * ENC + h * HD;
    const h16* vb = vt16 + ((size_t)b * NHEAD + h) * HD * NS;

    v16h qa0 = ldA(qb, ENC, cl, 0, lane);     // reused for all key tiles
    v16h qa1 = ldA(qb, ENC, cl, 32, lane);
    v8f acc[4] = {{}, {}, {}, {}};
    float mrow[8], lrow[8];
#pragma unroll
    for (int v = 0; v < 8; ++v) { mrow[v] = -3.0e38f; lrow[v] = 0.f; }

    for (int kt = 0; kt < NS; kt += 32) {
        v8f s0 = {}, s1 = {};
        s0 = __builtin_amdgcn_wmma_f32_16x16x32_f16(false, qa0, false, ldBT(kb, ENC, kt + cl, 0, lane), (short)0, s0, false, false);
        s0 = __builtin_amdgcn_wmma_f32_16x16x32_f16(false, qa1, false, ldBT(kb, ENC, kt + cl, 32, lane), (short)0, s0, false, false);
        s1 = __builtin_amdgcn_wmma_f32_16x16x32_f16(false, qa0, false, ldBT(kb, ENC, kt + 16 + cl, 0, lane), (short)0, s1, false, false);
        s1 = __builtin_amdgcn_wmma_f32_16x16x32_f16(false, qa1, false, ldBT(kb, ENC, kt + 16 + cl, 32, lane), (short)0, s1, false, false);
#pragma unroll
        for (int v = 0; v < 8; ++v) {
            float a0 = s0[v] * 0.125f, a1 = s1[v] * 0.125f;   // 1/sqrt(64)
            float tm = red16_max(fmaxf(a0, a1));
            float nm = fmaxf(mrow[v], tm);
            float alpha = __expf(mrow[v] - nm);
            float p0 = __expf(a0 - nm), p1 = __expf(a1 - nm);
            lrow[v] = lrow[v] * alpha + red16_sum(p0 + p1);
            mrow[v] = nm;
#pragma unroll
            for (int t = 0; t < 4; ++t) acc[t][v] *= alpha;
            ptile[wave][v + 8 * g][cl] = (h16)p0;
            ptile[wave][v + 8 * g][16 + cl] = (h16)p1;
        }
        asm volatile("s_wait_dscnt 0x0" ::: "memory");   // same-wave LDS write->read
        const h16* pp = &ptile[wave][cl][8 * g];
        v16h pa = cat8(*(const v8h*)pp, *(const v8h*)(pp + 16));
#pragma unroll
        for (int t = 0; t < 4; ++t) {
            v16h bf = ldBT(vb, NS, t * 16 + cl, kt, lane);
            acc[t] = __builtin_amdgcn_wmma_f32_16x16x32_f16(false, pa, false, bf, (short)0, acc[t], false, false);
        }
    }
#pragma unroll
    for (int t = 0; t < 4; ++t)
#pragma unroll
        for (int v = 0; v < 8; ++v) {
            int r = qt0 + v + 8 * g;
            o16[((size_t)(b * NS + r)) * ENC + h * HD + t * 16 + cl] = (h16)(acc[t][v] / lrow[v]);
        }
}

__global__ void copy_out_kernel(const float* __restrict__ x, float* __restrict__ out, int n) {
    int i = blockIdx.x * 256 + threadIdx.x;
    if (i < n) out[i] = x[i];
}

// ---------------------------------------------------------------- host
extern "C" void kernel_launch(void* const* d_in, const int* in_sizes, int n_in,
                              void* d_out, int out_size, void* d_ws, size_t ws_size,
                              hipStream_t stream) {
    (void)in_sizes; (void)n_in; (void)out_size; (void)ws_size;
    const float* points = (const float*)d_in[0];
    const int*   snidx  = (const int*)d_in[1];
    const float* pool_w = (const float*)d_in[2];
    const float* pool_b = (const float*)d_in[3];
    const float* msg_w1 = (const float*)d_in[4];
    const float* msg_b1 = (const float*)d_in[5];
    const float* msg_w2 = (const float*)d_in[6];
    const float* msg_b2 = (const float*)d_in[7];
    const float* proj_w = (const float*)d_in[8];
    const float* proj_b = (const float*)d_in[9];

    char* base = (char*)d_ws;
    size_t off = 0;
    auto alloc = [&](size_t bytes) -> void* {
        off = (off + 255) & ~(size_t)255;
        void* p = base + off; off += bytes; return p;
    };
    float* nbr    = (float*)alloc((size_t)TOK * KNBR * 3 * 4);
    h16*   pooled = (h16*)  alloc((size_t)TOK * GNN * 2);
    float* xenc   = (float*)alloc((size_t)TOK * ENC * 4);
    h16*   hb     = (h16*)  alloc((size_t)TOK * ENC * 2);
    h16*   qb     = (h16*)  alloc((size_t)TOK * ENC * 2);
    h16*   kb     = (h16*)  alloc((size_t)TOK * ENC * 2);
    h16*   vb     = (h16*)  alloc((size_t)TOK * ENC * 2);
    h16*   vtb    = (h16*)  alloc((size_t)TOK * ENC * 2);
    h16*   ob     = (h16*)  alloc((size_t)TOK * ENC * 2);
    h16*   mb     = (h16*)  alloc((size_t)TOK * MLPH * 2);
    h16*   w1T    = (h16*)  alloc((size_t)GNN * GNN * 2);
    h16*   w2T    = (h16*)  alloc((size_t)GNN * GNN * 2);
    h16*   pjT    = (h16*)  alloc((size_t)GNN * ENC * 2);
    h16 *wqT[DEPTH], *wkT[DEPTH], *wvT[DEPTH], *woT[DEPTH], *m1T[DEPTH], *m2T[DEPTH];
    for (int d = 0; d < DEPTH; ++d) {
        wqT[d] = (h16*)alloc((size_t)ENC * ENC * 2);
        wkT[d] = (h16*)alloc((size_t)ENC * ENC * 2);
        wvT[d] = (h16*)alloc((size_t)ENC * ENC * 2);
        woT[d] = (h16*)alloc((size_t)ENC * ENC * 2);
        m1T[d] = (h16*)alloc((size_t)ENC * MLPH * 2);
        m2T[d] = (h16*)alloc((size_t)MLPH * ENC * 2);
    }

    auto prep = [&](const float* W, h16* WT, int Kd, int N) {
        int tot = Kd * N;
        wprep_kernel<<<(tot + 255) / 256, 256, 0, stream>>>(W, WT, Kd, N);
    };
    prep(msg_w1, w1T, GNN, GNN);
    prep(msg_w2, w2T, GNN, GNN);
    prep(proj_w, pjT, GNN, ENC);
    for (int d = 0; d < DEPTH; ++d) {
        int bs = 10 + 16 * d;
        prep((const float*)d_in[bs + 2],  wqT[d], ENC, ENC);
        prep((const float*)d_in[bs + 4],  wkT[d], ENC, ENC);
        prep((const float*)d_in[bs + 6],  wvT[d], ENC, ENC);
        prep((const float*)d_in[bs + 8],  woT[d], ENC, ENC);
        prep((const float*)d_in[bs + 12], m1T[d], ENC, MLPH);
        prep((const float*)d_in[bs + 14], m2T[d], MLPH, ENC);
    }

    // 1) KNN gather
    knn_kernel<<<TOK / KNN_WAVES, 32 * KNN_WAVES, 0, stream>>>(points, snidx, nbr);
    // 2) fused GNN (pool + sincos + MLP + mean)
    gnn_kernel<<<TOK, 128, 0, stream>>>(nbr, pool_w, pool_b, w1T, msg_b1, w2T, msg_b2, pooled);
    // 3) projection to ENC (fp32 residual stream)
    gemm_kernel<<<dim3(ENC / 64, TOK / 128), 256, 0, stream>>>(
        pooled, pjT, proj_b, xenc, nullptr, nullptr, TOK, ENC, GNN, 0);
    // 4) transformer blocks
    for (int d = 0; d < DEPTH; ++d) {
        int bs = 10 + 16 * d;
        ln_kernel<<<TOK / 8, 256, 0, stream>>>(xenc, (const float*)d_in[bs + 0], (const float*)d_in[bs + 1], hb);
        gemm_kernel<<<dim3(ENC / 64, TOK / 128), 256, 0, stream>>>(
            hb, wqT[d], (const float*)d_in[bs + 3], nullptr, qb, nullptr, TOK, ENC, ENC, 0);
        gemm_kernel<<<dim3(ENC / 64, TOK / 128), 256, 0, stream>>>(
            hb, wkT[d], (const float*)d_in[bs + 5], nullptr, kb, nullptr, TOK, ENC, ENC, 0);
        gemm_kernel<<<dim3(ENC / 64, TOK / 128), 256, 0, stream>>>(
            hb, wvT[d], (const float*)d_in[bs + 7], nullptr, vb, nullptr, TOK, ENC, ENC, 0);
        vtr_kernel<<<(TOK * ENC + 255) / 256, 256, 0, stream>>>(vb, vtb);
        attn_kernel<<<dim3(NS / 64, NHEAD, NB), 128, 0, stream>>>(qb, kb, vtb, ob);
        gemm_kernel<<<dim3(ENC / 64, TOK / 128), 256, 0, stream>>>(
            ob, woT[d], (const float*)d_in[bs + 9], xenc, nullptr, xenc, TOK, ENC, ENC, 0);
        ln_kernel<<<TOK / 8, 256, 0, stream>>>(xenc, (const float*)d_in[bs + 10], (const float*)d_in[bs + 11], hb);
        gemm_kernel<<<dim3(MLPH / 64, TOK / 128), 256, 0, stream>>>(
            hb, m1T[d], (const float*)d_in[bs + 13], nullptr, mb, nullptr, TOK, MLPH, ENC, 1);
        gemm_kernel<<<dim3(ENC / 64, TOK / 128), 256, 0, stream>>>(
            mb, m2T[d], (const float*)d_in[bs + 15], xenc, nullptr, xenc, TOK, ENC, MLPH, 0);
    }
    // 5) outputs: x then points
    copy_out_kernel<<<(TOK * ENC + 255) / 256, 256, 0, stream>>>(xenc, (float*)d_out, TOK * ENC);
    hipMemcpyAsync((float*)d_out + (size_t)TOK * ENC, points,
                   (size_t)NB * NPTS * 3 * sizeof(float), hipMemcpyDeviceToDevice, stream);
}
